// MultiQueryAttention_31628139167856
// MI455X (gfx1250) — matrix operand order repaired and checked by reading, not compiled, not run on hardware
//
#include <hip/hip_runtime.h>
#include <hip/hip_bf16.h>

// ---------------------------------------------------------------------------
// MQA forward for MI455X (gfx1250, wave32).
//  - All matmuls: v_wmma_f32_16x16x32_f16 (f32 accumulate)
//  - GEMM B tile staged by the Tensor Data Mover (tensor_load_to_lds,
//    TENSORcnt), A tile by global_load_async_to_lds_b128 (ASYNCcnt),
//    double-buffered so DMA overlaps WMMA compute.
// ---------------------------------------------------------------------------

typedef __attribute__((ext_vector_type(16))) _Float16 v16h;
typedef __attribute__((ext_vector_type(8)))  _Float16 v8h;
typedef __attribute__((ext_vector_type(8)))  float    v8f;
typedef __attribute__((ext_vector_type(4)))  unsigned u32x4;
typedef __attribute__((ext_vector_type(8)))  unsigned u32x8;

union Frag16 { v16h v; v8h h[2]; };

#define HIDDEN   2048
#define NHEADS   16
#define NKV      2
#define HEADDIM  128
#define SEQ      2048
#define BATCH    2
#define MTOT     (BATCH * SEQ)      // 4096
#define KVDIM    (NKV * HEADDIM)    // 256

// ---------------------------------------------------------------------------
// A fragment: 16x32 f16 (M x K), row-major source with leading dim lda.
// Contiguous k order used here (B loads match): lane L holds row M=L%16; elements 0..15 = K[(L/16)*16 .. +16).
// ---------------------------------------------------------------------------
__device__ inline v16h load_a_frag(const _Float16* __restrict__ A, int lda,
                                   int row0, int k0, int lane) {
  int l = lane & 15, hh = lane >> 4;
  const _Float16* base = A + (size_t)(row0 + l) * lda + k0;
  Frag16 f;
  f.h[0] = *(const v8h*)(base + hh * 16);
  f.h[1] = *(const v8h*)(base + hh * 16 + 8);
  return f.v;
}

// ---------------------------------------------------------------------------
// TDM: DMA a 64-row x 32-element f16 tile (row stride = K elements) from
// global memory into LDS. Builds the D# descriptor (groups 0/1) in SGPRs;
// 2-D tensor -> VADDR2/VADDR3 omitted. Tracked by TENSORcnt.
// ---------------------------------------------------------------------------
__device__ inline void tdm_load_b_tile(const _Float16* gptr, unsigned lds_addr,
                                       int K, int N) {
  unsigned long long ga = (unsigned long long)(uintptr_t)gptr;
  u32x4 g0;
  g0[0] = 1u;                                    // count=1, user mode
  g0[1] = lds_addr;                              // LDS byte address
  g0[2] = (unsigned)ga;                          // global_addr[31:0]
  g0[3] = (unsigned)((ga >> 32) & 0x01FFFFFFull) // global_addr[56:32]
          | (2u << 30);                          // type=2 ("image")
  u32x8 g1;
  g1[0] = 1u << 16;                              // data_size=1 -> 2 bytes
  g1[1] = ((unsigned)K & 0xFFFFu) << 16;         // tensor_dim0[15:0]
  g1[2] = (((unsigned)K >> 16) & 0xFFFFu)        // tensor_dim0[31:16]
          | (((unsigned)N & 0xFFFFu) << 16);     // tensor_dim1[15:0]
  g1[3] = (((unsigned)N >> 16) & 0xFFFFu)        // tensor_dim1[31:16]
          | (32u << 16);                         // tile_dim0 = 32 elements
  g1[4] = 64u;                                   // tile_dim1 = 64 rows
  g1[5] = (unsigned)K;                           // tensor_dim0_stride[31:0]
  g1[6] = 0u;                                    // stride[47:32], dim1_stride
  g1[7] = 0u;
  asm volatile("tensor_load_to_lds %0, %1" : : "s"(g0), "s"(g1) : "memory");
}

// ---------------------------------------------------------------------------
// Elementwise f32 -> f16
// ---------------------------------------------------------------------------
__global__ void cvt_f32_to_f16_k(const float* __restrict__ in,
                                 _Float16* __restrict__ out, size_t n) {
  size_t i = (size_t)blockIdx.x * blockDim.x + threadIdx.x;
  if (i < n) out[i] = (_Float16)in[i];
}

// W [K][N] f32  ->  Wt [N][K] f16
__global__ void transpose_f32_to_f16_k(const float* __restrict__ in,
                                       _Float16* __restrict__ out,
                                       int K, int N) {
  size_t i = (size_t)blockIdx.x * blockDim.x + threadIdx.x;
  size_t n = (size_t)K * N;
  if (i >= n) return;
  int kk = (int)(i / N);
  int nn = (int)(i % N);
  out[(size_t)nn * K + kk] = (_Float16)in[i];
}

// v [B*S][KV*D] f16 -> vt [(b*KV + kv)*D + d][S] f16
__global__ void vtrans_k(const _Float16* __restrict__ v,
                         _Float16* __restrict__ vt) {
  size_t i = (size_t)blockIdx.x * blockDim.x + threadIdx.x;
  size_t n = (size_t)MTOT * KVDIM;
  if (i >= n) return;
  int col = (int)(i % KVDIM);          // kv*128 + d
  int row = (int)(i / KVDIM);          // b*S + s
  int b = row / SEQ, s = row % SEQ;
  vt[((size_t)b * KVDIM + col) * SEQ + s] = v[i];
}

// ---------------------------------------------------------------------------
// GEMM: C[M][N] = (A[M][K] @ B[K][N] + bias[N]) * scale
// A: f16 row-major. Bt: f16 [N][K] (B transposed). Block tile 128x64,
// 8 waves (256 threads), each wave owns a 16x64 strip (4 WMMA accumulators).
// Double-buffered LDS staging: B tile via TDM (wave 0, TENSORcnt), A tile
// via per-lane async global->LDS (ASYNCcnt). DMA of tile i+1 overlaps WMMA
// of tile i.
// ---------------------------------------------------------------------------
template <bool OUT_F32>
__global__ __launch_bounds__(256)
void gemm_wmma_k(const _Float16* __restrict__ A,
                 const _Float16* __restrict__ Bt,
                 const float* __restrict__ bias,
                 void* __restrict__ Cout,
                 int M, int N, int K, float scale) {
  __shared__ __align__(16) _Float16 Bs[2][64 * 32];    // 2 x 4 KB
  __shared__ __align__(16) _Float16 As[2][128 * 32];   // 2 x 8 KB

  int tid  = threadIdx.x;
  int lane = tid & 31;
  int wave = tid >> 5;                 // 0..7
  int l = lane & 15, hh = lane >> 4;

  int rowblk = blockIdx.y * 128;
  int row0   = rowblk + wave * 16;
  int n0     = blockIdx.x * 64;
  (void)row0;

  v8f acc[4] = {};

  // A staging: thread t moves 32 B of row (t>>1), half (t&1) of the k-chunk.
  int ar = tid >> 1;                   // 0..127
  int ac = (tid & 1) * 16;             // 0 or 16 (f16 elements)
  const _Float16* asrc = A + (size_t)(rowblk + ar) * K + ac;
  unsigned adst[2] = {
      (unsigned)(unsigned long long)(uintptr_t)&As[0][ar * 32 + ac],
      (unsigned)(unsigned long long)(uintptr_t)&As[1][ar * 32 + ac]};
  unsigned bdst[2] = {
      (unsigned)(unsigned long long)(uintptr_t)&Bs[0][0],
      (unsigned)(unsigned long long)(uintptr_t)&Bs[1][0]};

  auto issue_tile = [&](int bf, int k0) {
    // A tile: two b128 async copies (INST_OFFSET applies to LDS and global)
    unsigned long long gsrc = (unsigned long long)(uintptr_t)(asrc + k0);
    asm volatile(
        "global_load_async_to_lds_b128 %0, %1, off\n\t"
        "global_load_async_to_lds_b128 %0, %1, off offset:16"
        :
        : "v"(adst[bf]), "v"(gsrc)
        : "memory");
    // B tile: one TDM descriptor moves the whole 64x32 tile
    if (wave == 0)
      tdm_load_b_tile(Bt + (size_t)n0 * K + k0, bdst[bf], K, N);
  };

  issue_tile(0, 0);

  int bf = 0;
  for (int k0 = 0; k0 < K; k0 += 32) {
    if (wave == 0) __builtin_amdgcn_s_wait_tensorcnt(0);
    asm volatile("s_wait_asynccnt 0" ::: "memory");
    __syncthreads();
    if (k0 + 32 < K) issue_tile(bf ^ 1, k0 + 32);

    // A fragment from LDS
    Frag16 fa;
    {
      const _Float16* ab = &As[bf][(wave * 16 + l) * 32];
      fa.h[0] = *(const v8h*)(ab + hh * 16);
      fa.h[1] = *(const v8h*)(ab + hh * 16 + 8);
    }

#pragma unroll
    for (int c = 0; c < 4; ++c) {
      Frag16 bfr;
      const _Float16* bb = &Bs[bf][(c * 16 + l) * 32 + hh * 16];
      bfr.h[0] = *(const v8h*)(bb);
      bfr.h[1] = *(const v8h*)(bb + 8);
      acc[c] = __builtin_amdgcn_wmma_f32_16x16x32_f16(
          false, fa.v, false, bfr.v, (short)0, acc[c], false, false);
    }
    bf ^= 1;
  }

  // epilogue: C-layout element (r) -> row = row0 + r + 8*hh, col = n0+c*16+l
#pragma unroll
  for (int c = 0; c < 4; ++c) {
#pragma unroll
    for (int r = 0; r < 8; ++r) {
      int row = rowblk + wave * 16 + r + 8 * hh;
      int col = n0 + c * 16 + l;
      float vv = (acc[c][r] + bias[col]) * scale;
      if (OUT_F32)
        ((float*)Cout)[(size_t)row * N + col] = vv;
      else
        ((_Float16*)Cout)[(size_t)row * N + col] = (_Float16)vv;
    }
  }
}

// ---------------------------------------------------------------------------
// Flash-style MQA attention. One wave per (b, head, 16-query tile).
// Key tiles of 32 -> two 16x16 score accumulators; exact K=32 contraction
// for the P@V WMMA. P goes C-layout -> A-layout through a 1KB LDS tile
// (same-wave DS ops are in-order). Row reductions with half-wave shuffles.
//   q : [B*S][H*D] f16, pre-scaled by 1/sqrt(D)
//   k : [B*S][KV*D] f16
//   vt: [(b*KV+kv)*D + d][S] f16
//   o : [B*S][H*D] f16
// ---------------------------------------------------------------------------
__global__ __launch_bounds__(32)
void mqa_attn_k(const _Float16* __restrict__ q,
                const _Float16* __restrict__ k,
                const _Float16* __restrict__ vt,
                _Float16* __restrict__ o) {
  __shared__ __align__(16) _Float16 Ps[16 * 32];

  int qt = blockIdx.x;               // 0..S/16-1
  int h  = blockIdx.y;               // 0..15
  int b  = blockIdx.z;               // 0..1
  int kv = h >> 3;                   // rep = NHEADS/NKV = 8 (repeat_interleave)
  int lane = threadIdx.x & 31;
  int l = lane & 15, hh = lane >> 4;
  int q0 = qt * 16;

  // Q fragments: 16 queries x 128 d, 4 chunks of K=32
  v16h qf[4];
#pragma unroll
  for (int c = 0; c < 4; ++c)
    qf[c] = load_a_frag(q, NHEADS * HEADDIM, b * SEQ + q0,
                        h * HEADDIM + c * 32, lane);

  float m[8], lsum[8];
  v8f o_acc[8];
#pragma unroll
  for (int r = 0; r < 8; ++r) { m[r] = -1e30f; lsum[r] = 0.0f; }
#pragma unroll
  for (int dc = 0; dc < 8; ++dc) { v8f z = {}; o_acc[dc] = z; }

  for (int kt = 0; kt < SEQ; kt += 32) {
    // ---- scores: S0 = Q @ K[kt..kt+16)^T, S1 = Q @ K[kt+16..kt+32)^T ----
    v8f s0 = {}, s1 = {};
#pragma unroll
    for (int c = 0; c < 4; ++c) {
      const _Float16* kb0 =
          k + ((size_t)(b * SEQ + kt + l) * NKV + kv) * HEADDIM
            + c * 32 + hh * 16;
      const _Float16* kb1 =
          k + ((size_t)(b * SEQ + kt + 16 + l) * NKV + kv) * HEADDIM
            + c * 32 + hh * 16;
      Frag16 kf0, kf1;
      kf0.h[0] = *(const v8h*)kb0; kf0.h[1] = *(const v8h*)(kb0 + 8);
      kf1.h[0] = *(const v8h*)kb1; kf1.h[1] = *(const v8h*)(kb1 + 8);
      s0 = __builtin_amdgcn_wmma_f32_16x16x32_f16(
          false, qf[c], false, kf0.v, (short)0, s0, false, false);
      s1 = __builtin_amdgcn_wmma_f32_16x16x32_f16(
          false, qf[c], false, kf1.v, (short)0, s1, false, false);
    }

    // ---- online softmax (per-row, rows live across 16-lane halves) ----
    float sc[8];
#pragma unroll
    for (int r = 0; r < 8; ++r) {
      float local = fmaxf(s0[r], s1[r]);
#pragma unroll
      for (int w = 8; w >= 1; w >>= 1)
        local = fmaxf(local, __shfl_xor(local, w, 16));
      float mnew = fmaxf(m[r], local);
      sc[r] = __expf(m[r] - mnew);
      float p0 = __expf(s0[r] - mnew);
      float p1 = __expf(s1[r] - mnew);
      float rs = p0 + p1;
#pragma unroll
      for (int w = 8; w >= 1; w >>= 1)
        rs += __shfl_xor(rs, w, 16);
      lsum[r] = lsum[r] * sc[r] + rs;
      m[r] = mnew;
      // C-layout -> LDS [16 rows][32 keys]
      Ps[(r + 8 * hh) * 32 + l]      = (_Float16)p0;
      Ps[(r + 8 * hh) * 32 + 16 + l] = (_Float16)p1;
    }

    // rescale running output
#pragma unroll
    for (int dc = 0; dc < 8; ++dc) {
#pragma unroll
      for (int r = 0; r < 8; ++r) o_acc[dc][r] *= sc[r];
    }

    // ---- P as A-operand (16x32) from LDS (same-wave DS is in-order) ----
    Frag16 pf;
    {
      const _Float16* pb = &Ps[l * 32];
      pf.h[0] = *(const v8h*)(pb + hh * 16);
      pf.h[1] = *(const v8h*)(pb + hh * 16 + 8);
    }

    // ---- O += P @ V, 8 chunks of 16 output columns ----
#pragma unroll
    for (int dc = 0; dc < 8; ++dc) {
      const _Float16* vb =
          vt + ((size_t)((b * NKV + kv) * HEADDIM + dc * 16 + l)) * SEQ
             + kt + hh * 16;
      Frag16 vf;
      vf.h[0] = *(const v8h*)vb; vf.h[1] = *(const v8h*)(vb + 8);
      o_acc[dc] = __builtin_amdgcn_wmma_f32_16x16x32_f16(
          false, pf.v, false, vf.v, (short)0, o_acc[dc], false, false);
    }
  }

  // ---- normalize and write out ----
#pragma unroll
  for (int r = 0; r < 8; ++r) {
    float inv = 1.0f / lsum[r];
    int row = b * SEQ + q0 + r + 8 * hh;
#pragma unroll
    for (int dc = 0; dc < 8; ++dc) {
      o[(size_t)row * (NHEADS * HEADDIM) + h * HEADDIM + dc * 16 + l] =
          (_Float16)(o_acc[dc][r] * inv);
    }
  }
}

// ---------------------------------------------------------------------------
// Host-side orchestration
// ---------------------------------------------------------------------------
extern "C" void kernel_launch(void* const* d_in, const int* in_sizes, int n_in,
                              void* d_out, int out_size, void* d_ws,
                              size_t ws_size, hipStream_t stream) {
  const float* hs = (const float*)d_in[0];
  const float* Wq = (const float*)d_in[1];
  const float* bq = (const float*)d_in[2];
  const float* Wk = (const float*)d_in[3];
  const float* bk = (const float*)d_in[4];
  const float* Wv = (const float*)d_in[5];
  const float* bv = (const float*)d_in[6];
  const float* Wo = (const float*)d_in[7];
  const float* bo = (const float*)d_in[8];
  float* out = (float*)d_out;

  char* ws = (char*)d_ws;
  size_t off = 0;
  _Float16* Hh  = (_Float16*)(ws + off); off += (size_t)MTOT * HIDDEN * 2;   // 16 MB
  _Float16* Wqt = (_Float16*)(ws + off); off += (size_t)HIDDEN * HIDDEN * 2; //  8 MB
  _Float16* Wkt = (_Float16*)(ws + off); off += (size_t)KVDIM * HIDDEN * 2;  //  1 MB
  _Float16* Wvt = (_Float16*)(ws + off); off += (size_t)KVDIM * HIDDEN * 2;  //  1 MB
  _Float16* Wot = (_Float16*)(ws + off); off += (size_t)HIDDEN * HIDDEN * 2; //  8 MB
  _Float16* Qh  = (_Float16*)(ws + off); off += (size_t)MTOT * HIDDEN * 2;   // 16 MB
  _Float16* Kh  = (_Float16*)(ws + off); off += (size_t)MTOT * KVDIM * 2;    //  2 MB
  _Float16* Vh  = (_Float16*)(ws + off); off += (size_t)MTOT * KVDIM * 2;    //  2 MB
  _Float16* Vt  = (_Float16*)(ws + off); off += (size_t)MTOT * KVDIM * 2;    //  2 MB
  _Float16* Ah  = (_Float16*)(ws + off); off += (size_t)MTOT * HIDDEN * 2;   // 16 MB

  // 1) down-convert activations + transpose weights to f16
  {
    size_t n = (size_t)MTOT * HIDDEN;
    cvt_f32_to_f16_k<<<(unsigned)((n + 255) / 256), 256, 0, stream>>>(hs, Hh, n);
  }
  {
    size_t n = (size_t)HIDDEN * HIDDEN;
    transpose_f32_to_f16_k<<<(unsigned)((n + 255) / 256), 256, 0, stream>>>(
        Wq, Wqt, HIDDEN, HIDDEN);
    transpose_f32_to_f16_k<<<(unsigned)((n + 255) / 256), 256, 0, stream>>>(
        Wo, Wot, HIDDEN, HIDDEN);
  }
  {
    size_t n = (size_t)HIDDEN * KVDIM;
    transpose_f32_to_f16_k<<<(unsigned)((n + 255) / 256), 256, 0, stream>>>(
        Wk, Wkt, HIDDEN, KVDIM);
    transpose_f32_to_f16_k<<<(unsigned)((n + 255) / 256), 256, 0, stream>>>(
        Wv, Wvt, HIDDEN, KVDIM);
  }

  // 2) projections (q pre-scaled by 1/sqrt(HEAD_DIM))
  dim3 blk(256);
  dim3 gq(HIDDEN / 64, MTOT / 128);
  gemm_wmma_k<false><<<gq, blk, 0, stream>>>(Hh, Wqt, bq, Qh, MTOT, HIDDEN,
                                             HIDDEN, 0.08838834764831845f);
  dim3 gkv(KVDIM / 64, MTOT / 128);
  gemm_wmma_k<false><<<gkv, blk, 0, stream>>>(Hh, Wkt, bk, Kh, MTOT, KVDIM,
                                              HIDDEN, 1.0f);
  gemm_wmma_k<false><<<gkv, blk, 0, stream>>>(Hh, Wvt, bv, Vh, MTOT, KVDIM,
                                              HIDDEN, 1.0f);

  // 3) transpose V for the PV matmul (contiguous along keys)
  {
    size_t n = (size_t)MTOT * KVDIM;
    vtrans_k<<<(unsigned)((n + 255) / 256), 256, 0, stream>>>(Vh, Vt);
  }

  // 4) attention
  dim3 ga(SEQ / 16, NHEADS, BATCH);
  mqa_attn_k<<<ga, 32, 0, stream>>>(Qh, Kh, Vt, Ah);

  // 5) output projection -> f32 result
  gemm_wmma_k<true><<<gq, blk, 0, stream>>>(Ah, Wot, bo, out, MTOT, HIDDEN,
                                            HIDDEN, 1.0f);
}
